// QuantumLSTM_27522150433023
// MI455X (gfx1250) — matrix operand order repaired and checked by reading, not compiled, not run on hardware
//
#include <hip/hip_runtime.h>
#include <hip/hip_bf16.h>
#include <math.h>

// Quantum-LSTM for MI455X (gfx1250), round 2.
//
// Reformulation: the circuit after RX AngleEmbedding is a FIXED 64x64 complex
// unitary per LSTM gate (weights are constant per call).  The embedded state
// is a product state  amp[k] = (-i)^popcount(k) * p_k  (p_k = product of 6
// cos/sin terms), so with A'[k] = sigma(k)*p_k (sigma = -1 iff popcount mod 4
// in {2,3}) and parity-permuted unitary rows
//     Br[k] = pc even ?  Gre[k] : Gim[k]
//     Bi[k] = pc even ?  Gim[k] : -Gre[k]
// we get   Fr = A'@Br,  Fi = A'@Bi   -- two real K=64 GEMMs (4 WMMAs/N-tile).
// PauliZ expvals + W_out are fused:  z = probs @ (Z * W_out^T) + b_out,
// one more K=64 WMMA pair against a fixed +-signed matrix ZW.
//
// Main kernel: 128 blocks x 128 threads (4 waves); wave = LSTM gate; block
// owns 16 batch rows for all T=64 steps (rows independent -> no grid sync).

#define T_LEN 64
#define N_IN  8
#define N_HID 6      // qubits
#define N_OUT 8
#define NST   64     // 2^6 amplitudes
#define NGATE 4

typedef __attribute__((ext_vector_type(16))) _Float16 v16h;
typedef __attribute__((ext_vector_type(8)))  float    v8f;

static __device__ __forceinline__ v8f wmma_f16(v16h a, v16h b, v8f c) {
  // D = A(16x32,f16) * B(32x16,f16) + C(16x16,f32)
  return __builtin_amdgcn_wmma_f32_16x16x32_f16(false, a, false, b,
                                                (short)0, c, false, false);
}

// ---------------------------------------------------------------------------
// Setup: thread (g,k) evolves basis state |k> through gate g's fixed layers,
// giving G[k][n] = U_g[n][k]; stores parity-folded rows (Br, Bi) as float2.
// ---------------------------------------------------------------------------
__global__ void __launch_bounds__(256)
vqc_build_unitaries(const float* __restrict__ wf, const float* __restrict__ wi,
                    const float* __restrict__ wg, const float* __restrict__ wo,
                    float* __restrict__ Gpack /* [4][64][64][2] = {Br,Bi} */) {
  const int tid = threadIdx.x;     // 0..255
  const int g   = tid >> 6;        // gate id
  const int k   = tid & 63;        // basis column
  const float* W = (g == 0) ? wf : (g == 1) ? wi : (g == 2) ? wg : wo;

  float re[NST], im[NST];
#pragma unroll
  for (int i = 0; i < NST; ++i) { re[i] = 0.0f; im[i] = 0.0f; }
  re[k] = 1.0f;

#pragma unroll
  for (int l = 0; l < 2; ++l) {
    // RX(theta) on wire w  (wire w <-> bit (5-w))
#pragma unroll
    for (int w = 0; w < N_HID; ++w) {
      const float th = W[l * N_HID + w] * 0.5f;
      const float cc = cosf(th), ss = sinf(th);
      const int mask = 1 << (5 - w);
#pragma unroll
      for (int i = 0; i < NST; ++i) {
        if (i & mask) continue;
        const int j = i | mask;
        const float a0r = re[i], a0i = im[i], a1r = re[j], a1i = im[j];
        re[i] = cc * a0r + ss * a1i;   // b0 = c*a0 - i*s*a1
        im[i] = cc * a0i - ss * a1r;
        re[j] = cc * a1r + ss * a0i;   // b1 = -i*s*a0 + c*a1
        im[j] = cc * a1i - ss * a0r;
      }
    }
    // CNOT ring (w, (w+1)%6)
#pragma unroll
    for (int w = 0; w < N_HID; ++w) {
      const int mc = 1 << (5 - w);
      const int mt = 1 << (5 - ((w + 1) % N_HID));
#pragma unroll
      for (int i = 0; i < NST; ++i) {
        if ((i & mc) && !(i & mt)) {
          const int j = i | mt;
          const float tr = re[i], ti = im[i];
          re[i] = re[j]; im[i] = im[j];
          re[j] = tr;    im[j] = ti;
        }
      }
    }
  }
  // Parity-folded rows: row k of Br/Bi.
  const int par = __popc(k) & 1;
#pragma unroll
  for (int n = 0; n < NST; ++n) {
    const int idx = ((g * NST + k) * NST + n) * 2;
    Gpack[idx]     = par ? im[n]  : re[n];   // Br
    Gpack[idx + 1] = par ? -re[n] : im[n];   // Bi
  }
}

// ---------------------------------------------------------------------------
// Main kernel
// ---------------------------------------------------------------------------
__global__ void __launch_bounds__(128)
qlstm_main(const float* __restrict__ x, const float* __restrict__ W_in,
           const float* __restrict__ b_in, const float* __restrict__ W_out,
           const float* __restrict__ b_out, const float* __restrict__ Gpack,
           float* __restrict__ out) {
  const int lane   = threadIdx.x & 31;
  const int wave   = threadIdx.x >> 5;   // LSTM gate id (f,i,g,o)
  const int row    = lane & 15;          // batch row in tile (A-layout M)
  const int laneHi = lane >> 4;          // 0/1 (A-layout K+8, D-layout M+8)
  const int tileBase = blockIdx.x * 16;

  __shared__ float    sh_h[16][N_OUT];
  __shared__ float    sh_z[NGATE][16][N_OUT];
  __shared__ _Float16 sh_probs16[NGATE][16][NST];   // 8 KB
  __shared__ float    sh_Win[N_HID * 16];
  __shared__ float    sh_bin[N_HID];
  __shared__ float    sh_bout[N_OUT];

  for (int i = threadIdx.x; i < N_HID * 16; i += 128) sh_Win[i] = W_in[i];
  if (threadIdx.x < N_HID) sh_bin[threadIdx.x] = b_in[threadIdx.x];
  if (threadIdx.x < N_OUT) sh_bout[threadIdx.x] = b_out[threadIdx.x];
  if (wave == 0 && lane < 16) {
#pragma unroll
    for (int o = 0; o < N_OUT; ++o) sh_h[lane][o] = 0.0f;
  }
  float c8[N_OUT];                       // cell state (wave0, lanes<16)
#pragma unroll
  for (int o = 0; o < N_OUT; ++o) c8[o] = 0.0f;

  // Resident B operands (parity-folded unitary), B layout: lane%16 = N,
  // element h = K (table order).  n = nt*16 + row, k = kc*32 + kk(h, laneHi).
  v16h br[2][4], bi[2][4];
#pragma unroll
  for (int kc = 0; kc < 2; ++kc) {
#pragma unroll
    for (int nt = 0; nt < 4; ++nt) {
#pragma unroll
      for (int h = 0; h < 16; ++h) {
        const int k = kc * 32 + ((h < 8) ? h : h + 8) + laneHi * 8;
        const float* gp = Gpack + (size_t)((wave * NST + k) * NST) * 2;
        const int n = nt * 16 + row;
        br[kc][nt][h] = (_Float16)gp[2 * n];
        bi[kc][nt][h] = (_Float16)gp[2 * n + 1];
      }
    }
  }

  // ZW = Z(64x6, +-1) @ W_out^T(6x8), zero-padded to 16 cols; B layout:
  // lane%16 = output o, element h = K (state index n, table order).
  v16h zw[2];
#pragma unroll
  for (int kc = 0; kc < 2; ++kc) {
#pragma unroll
    for (int h = 0; h < 16; ++h) {
      const int n = kc * 32 + ((h < 8) ? h : h + 8) + laneHi * 8;
      float acc = 0.0f;
      if (row < 8) {
#pragma unroll
        for (int w = 0; w < N_HID; ++w) {
          const float wv = W_out[row * N_HID + w];
          acc += ((n >> (5 - w)) & 1) ? -wv : wv;
        }
      }
      zw[kc][h] = (_Float16)acc;
    }
  }
  __syncthreads();

#pragma unroll 1
  for (int t = 0; t < T_LEN; ++t) {
    // ---- angles y = W_in @ [h, x_t] + b_in for this lane's row ----
    const int b = tileBase + row;
    float v[16];
#pragma unroll
    for (int o = 0; o < N_OUT; ++o) v[o] = sh_h[row][o];
    const float* xp = x + (size_t)(b * T_LEN + t) * N_IN;
#pragma unroll
    for (int j = 0; j < N_IN; ++j) v[N_OUT + j] = xp[j];
    float cw[N_HID], sw[N_HID];
#pragma unroll
    for (int w = 0; w < N_HID; ++w) {
      float y = sh_bin[w];
#pragma unroll
      for (int j = 0; j < 16; ++j) y += sh_Win[w * 16 + j] * v[j];
      const float ha = 0.5f * y;
      cw[w] = cosf(ha);
      sw[w] = sinf(ha);
    }

    // ---- signed magnitudes A'[k] = sigma(k) * prod trig, A layout ----
    // kp bits: 5=kc(w0) 4=h>=8(w1) 3=laneHi(w2) 2..0=h&7(w3,w4,w5)
    const float m45_0 = cw[4] * cw[5], m45_1 = cw[4] * sw[5];
    const float m45_2 = sw[4] * cw[5], m45_3 = sw[4] * sw[5];
    const float m45[4] = {m45_0, m45_1, m45_2, m45_3};
    float tl[8];
#pragma unroll
    for (int j = 0; j < 8; ++j)
      tl[j] = ((j & 4) ? sw[3] : cw[3]) * m45[j & 3];
    const float z2 = laneHi ? sw[2] : cw[2];
    const float thv[4] = {cw[0] * cw[1] * z2, cw[0] * sw[1] * z2,
                          sw[0] * cw[1] * z2, sw[0] * sw[1] * z2};
    v16h ap[2];
#pragma unroll
    for (int kc = 0; kc < 2; ++kc) {
#pragma unroll
      for (int h = 0; h < 16; ++h) {
        const int b4  = (h >= 8) ? 1 : 0;
        const int lo3 = h & 7;
        const float p = thv[(kc << 1) | b4] * tl[lo3];
        const int pc_ct = kc + b4 + __popc(lo3);          // compile-time part
        const int neg = ((pc_ct + laneHi) >> 1) & 1;      // sigma sign
        ap[kc][h] = (_Float16)(neg ? -p : p);
      }
    }

    // ---- Fr = A'@Br, Fi = A'@Bi; probs = Fr^2+Fi^2 -> LDS (f16) ----
#pragma unroll
    for (int nt = 0; nt < 4; ++nt) {
      v8f fr = {};
      fr = wmma_f16(ap[0], br[0][nt], fr);
      fr = wmma_f16(ap[1], br[1][nt], fr);
      v8f fi = {};
      fi = wmma_f16(ap[0], bi[0][nt], fi);
      fi = wmma_f16(ap[1], bi[1][nt], fi);
#pragma unroll
      for (int r = 0; r < 8; ++r) {
        const float pr = fr[r] * fr[r] + fi[r] * fi[r];
        sh_probs16[wave][r + 8 * laneHi][nt * 16 + row] = (_Float16)pr;
      }
    }
    __syncthreads();

    // ---- fused z = probs @ ZW (+ b_out): reload probs in A layout ----
    v16h pa[2];
#pragma unroll
    for (int kc = 0; kc < 2; ++kc) {
#pragma unroll
      for (int h = 0; h < 16; ++h) {
        const int kk = ((h < 8) ? h : h + 8) + laneHi * 8 + kc * 32;
        pa[kc][h] = sh_probs16[wave][row][kk];
      }
    }
    v8f zz = {};
    zz = wmma_f16(pa[0], zw[0], zz);
    zz = wmma_f16(pa[1], zw[1], zz);
    if (row < 8) {                       // D layout: M=r+8*laneHi, o=lane%16
#pragma unroll
      for (int r = 0; r < 8; ++r)
        sh_z[wave][r + 8 * laneHi][row] = zz[r] + sh_bout[row];
    }
    __syncthreads();

    // ---- LSTM combine (wave 0, one lane per row) ----
    if (wave == 0 && lane < 16) {
      float* op = out + (size_t)((tileBase + lane) * T_LEN + t) * N_OUT;
#pragma unroll
      for (int o = 0; o < N_OUT; ++o) {
        const float fg = 1.0f / (1.0f + expf(-sh_z[0][lane][o]));
        const float ig = 1.0f / (1.0f + expf(-sh_z[1][lane][o]));
        const float gg = tanhf(sh_z[2][lane][o]);
        const float og = 1.0f / (1.0f + expf(-sh_z[3][lane][o]));
        c8[o] = fg * c8[o] + ig * gg;
        const float hv = og * tanhf(c8[o]);
        sh_h[lane][o] = hv;
        op[o] = hv;
      }
    }
    __syncthreads();
  }
}

// ---------------------------------------------------------------------------
extern "C" void kernel_launch(void* const* d_in, const int* in_sizes, int n_in,
                              void* d_out, int out_size, void* d_ws,
                              size_t ws_size, hipStream_t stream) {
  const float* x     = (const float*)d_in[0];
  const float* W_in  = (const float*)d_in[1];
  const float* b_in  = (const float*)d_in[2];
  const float* W_out = (const float*)d_in[3];
  const float* b_out = (const float*)d_in[4];
  const float* wf    = (const float*)d_in[5];
  const float* wi    = (const float*)d_in[6];
  const float* wg    = (const float*)d_in[7];
  const float* wo    = (const float*)d_in[8];
  float* Gpack = (float*)d_ws;    // needs 4*64*64*2*4 = 128 KB scratch
  float* outp  = (float*)d_out;   // [B, T, OUT] f32

  vqc_build_unitaries<<<1, 256, 0, stream>>>(wf, wi, wg, wo, Gpack);
  qlstm_main<<<128, 128, 0, stream>>>(x, W_in, b_in, W_out, b_out, Gpack,
                                      outp);
}
